// SkyerSdpaAttention_22900765622712
// MI455X (gfx1250) — matrix operand order repaired and checked by reading, not compiled, not run on hardware
//
#include <hip/hip_runtime.h>
#include <hip/hip_bf16.h>

// Problem constants (from reference): B=4, S=2048, D=2048, NQ=16, NKV=4, HD=128
#define B_   4
#define S_   2048
#define D_   2048
#define NQ_  16
#define NKV_ 4
#define HD_  128

typedef __attribute__((ext_vector_type(16))) __bf16 bf16x16;
typedef __attribute__((ext_vector_type(8)))  __bf16 bf16x8;
typedef __attribute__((ext_vector_type(8)))  float  f32x8;
typedef __attribute__((ext_vector_type(4)))  int    i32x4;

#define AS1 __attribute__((address_space(1)))
#define AS3 __attribute__((address_space(3)))

// CDNA5 async global->LDS copy path (ASYNCcnt-tracked), with sync fallback so
// the source compiles on toolchains lacking the builtins.
#if defined(__AMDGCN__) && __has_builtin(__builtin_amdgcn_global_load_async_to_lds_b128)
#define HAVE_ASYNC_LDS 1
#else
#define HAVE_ASYNC_LDS 0
#endif

static __device__ __forceinline__ void cp16(void* lds, const void* g) {
#if HAVE_ASYNC_LDS
  __builtin_amdgcn_global_load_async_to_lds_b128(
      (AS1 i32x4*)(unsigned long long)g,
      (AS3 i32x4*)(unsigned long long)lds, 0, 0);
#else
  *(bf16x8*)lds = *(const bf16x8*)g;
#endif
}

#if HAVE_ASYNC_LDS
#if defined(__AMDGCN__) && __has_builtin(__builtin_amdgcn_s_wait_asynccnt)
#define ASYNC_WAIT(n) __builtin_amdgcn_s_wait_asynccnt(n)
#else
#define ASYNC_WAIT(n) asm volatile("s_wait_asynccnt %0" ::"i"(n) : "memory")
#endif
#else
#define ASYNC_WAIT(n)
#endif

static __device__ __forceinline__ bf16x16 cat8(bf16x8 lo, bf16x8 hi) {
  bf16x16 r;
#pragma unroll
  for (int i = 0; i < 8; ++i) { r[i] = lo[i]; r[i + 8] = hi[i]; }
  return r;
}

static __device__ __forceinline__ f32x8 wmma_bf16(bf16x16 a, bf16x16 b, f32x8 c) {
  // D = A(16x32 bf16) * B(32x16 bf16) + C(16x16 f32)
  return __builtin_amdgcn_wmma_f32_16x16x32_bf16(false, a, false, b, (short)0, c,
                                                 false, false);
}

// ---------------------------------------------------------------------------
// fp32 -> bf16 conversion (plain and transposing variants)
// ---------------------------------------------------------------------------
__global__ __launch_bounds__(256)
void f2bf_kernel(const float* __restrict__ src, __bf16* __restrict__ dst, int n) {
  int i = blockIdx.x * 256 + threadIdx.x;
  if (i < n) dst[i] = (__bf16)src[i];
}

// src: [K][N] row-major fp32  ->  dst: [N][K] row-major bf16 (one-time cost)
__global__ __launch_bounds__(256)
void f2bfT_kernel(const float* __restrict__ src, __bf16* __restrict__ dst,
                  int K, int N) {
  int i = blockIdx.x * 256 + threadIdx.x;  // i = n*K + k -> coalesced writes
  if (i < K * N) {
    int n = i / K, k = i % K;
    dst[i] = (__bf16)src[(size_t)k * N + n];
  }
}

// ---------------------------------------------------------------------------
// Tiled WMMA GEMM:  C(MxN) = A(MxK bf16, row-major) @ B, with B supplied
// TRANSPOSED as BT(NxK bf16, row-major) so all LDS staging is contiguous.
// mode 0: outF[m*N+n] = C                       (final O @ Wo -> d_out fp32)
// mode 1: outH[b,h,s,d] = bf16(rope(C + bias))  (Q / K paths)
// mode 3: outH[b,kv,d,s] = bf16(C + bias)       (V path, d-major for flash)
// Block: 256 threads = 8 waves; tile 128x128, K-step 32, double-buffered
// async global->LDS copies.
// ---------------------------------------------------------------------------
#define GTM 128
#define GTN 128
#define GTK 32

__global__ __launch_bounds__(256)
void gemm_bf16_kernel(const __bf16* __restrict__ A, const __bf16* __restrict__ BT,
                      const float* __restrict__ bias, const float* __restrict__ fc,
                      const int* __restrict__ spv,
                      float* __restrict__ outF, __bf16* __restrict__ outH,
                      int M, int N, int K, int mode, int nheads) {
  // 64 KB LDS: phase1 = 2 x (As 8KB + BsT 8KB) double buffer; phase2 = Cs 64KB.
  __shared__ __align__(16) unsigned char smem[GTM * GTN * 4];
  float* Cs = (float*)smem;

  const int tid  = threadIdx.x;
  const int lane = tid & 31;
  const int wave = tid >> 5;
  const int wm   = wave >> 1;            // 0..3 : 32-row strip
  const int wn   = wave & 1;             // 0..1 : 64-col strip
  const int m0   = blockIdx.y * GTM;
  const int n0   = blockIdx.x * GTN;
  const int l15  = lane & 15;
  const int klo  = (lane < 16) ? 0 : 8;  // A-fragment K pattern (16-bit, K=32)
  const int kb   = (lane < 16) ? 0 : 8;  // B-fragment K pattern

  auto bufA = [&](int b) { return (__bf16*)(smem + b * 16384); };
  auto bufB = [&](int b) { return (__bf16*)(smem + b * 16384 + 8192); };

  // 4 async 16B copies per thread per tile (2 for A, 2 for BT).
  auto issue = [&](int k0, int b) {
    __bf16* Asb = bufA(b);
    __bf16* Bsb = bufB(b);
#pragma unroll
    for (int it = 0; it < 2; ++it) {
      int idx = tid + it * 256;          // 0..511
      int row = idx >> 1;                // 0..255
      int seg = (idx & 1) << 3;          // 0 or 8
      cp16(Asb + row * GTK + seg, A  + (size_t)(m0 + row) * K + k0 + seg);
      cp16(Bsb + row * GTK + seg, BT + (size_t)(n0 + row) * K + k0 + seg);
    }
  };

  f32x8 acc[2][4] = {};
  const int nk = K / GTK;
  issue(0, 0);
  for (int t = 0; t < nk; ++t) {
    const int cur = t & 1;
    if (t + 1 < nk) {
      issue((t + 1) * GTK, cur ^ 1);
      ASYNC_WAIT(4);                     // drain tile t, keep t+1 in flight
    } else {
      ASYNC_WAIT(0);
    }
    __syncthreads();

    const __bf16* Asb = bufA(cur);
    const __bf16* Bsb = bufB(cur);
    bf16x16 af[2];
#pragma unroll
    for (int mi = 0; mi < 2; ++mi) {
      int row = wm * 32 + mi * 16 + l15;
      af[mi] = cat8(*(const bf16x8*)(Asb + row * GTK + klo),
                    *(const bf16x8*)(Asb + row * GTK + klo + 16));
    }
#pragma unroll
    for (int ni = 0; ni < 4; ++ni) {
      int col = wn * 64 + ni * 16 + l15;
      bf16x16 bf = cat8(*(const bf16x8*)(Bsb + col * GTK + kb),
                        *(const bf16x8*)(Bsb + col * GTK + kb + 16));
#pragma unroll
      for (int mi = 0; mi < 2; ++mi)
        acc[mi][ni] = wmma_bf16(af[mi], bf, acc[mi][ni]);
    }
    __syncthreads();
  }

  // Spill accumulators to LDS so the epilogue can address arbitrarily.
  const int roff = (lane < 16) ? 0 : 8;  // C/D fragment: row = r + roff
#pragma unroll
  for (int mi = 0; mi < 2; ++mi)
#pragma unroll
    for (int ni = 0; ni < 4; ++ni)
#pragma unroll
      for (int r = 0; r < 8; ++r)
        Cs[(wm * 32 + mi * 16 + roff + r) * GTN + wn * 64 + ni * 16 + l15] =
            acc[mi][ni][r];
  __syncthreads();

  if (mode == 0) {
    for (int idx = tid; idx < GTM * GTN; idx += 256) {
      int i = idx >> 7;
      int j = idx & 127;
      outF[(size_t)(m0 + i) * N + n0 + j] = Cs[idx];
    }
  } else {
    const int sp0 = spv ? spv[0] : 0;
    for (int idx = tid; idx < GTM * GTN / 2; idx += 256) {
      int i  = idx >> 6;                 // row in tile
      int jp = idx & 63;                 // pair index
      int j  = jp * 2;
      int m  = m0 + i;
      int n  = n0 + j;
      float v0 = Cs[i * GTN + j]     + bias[n];
      float v1 = Cs[i * GTN + j + 1] + bias[n + 1];
      int bb = m / S_, ss = m % S_;
      int hh = n / HD_, dd = n % HD_;
      if (mode == 1) {  // RoPE, head-major [b,h,s,d]
        const float* f = fc + ((size_t)(ss + sp0) * (HD_ / 2) + (dd >> 1)) * 2;
        float cs = f[0], sn = f[1];
        size_t off = (((size_t)bb * nheads + hh) * S_ + ss) * (size_t)HD_ + dd;
        outH[off]     = (__bf16)(v0 * cs - v1 * sn);
        outH[off + 1] = (__bf16)(v0 * sn + v1 * cs);
      } else {          // V path, d-major [b,kv,d,s] so flash can async-copy V^T
        size_t off = (((size_t)bb * nheads + hh) * (size_t)HD_ + dd) * S_ + ss;
        outH[off]      = (__bf16)v0;
        outH[off + S_] = (__bf16)v1;
      }
    }
  }
}

// ---------------------------------------------------------------------------
// Flash attention (causal, GQA). Block = 256 threads = 8 waves; each wave owns
// 16 query rows of one (b, h); block covers 128 query rows. K (row-major) and
// V^T (d-major) tiles double-buffered in LDS via async copies. P restaged
// per-wave in LDS for the D->A fragment reshuffle.
// ---------------------------------------------------------------------------
__global__ __launch_bounds__(256)
void flash_kernel(const __bf16* __restrict__ Q, const __bf16* __restrict__ Kh,
                  const __bf16* __restrict__ VTh, __bf16* __restrict__ O) {
  __shared__ __align__(16) __bf16 Ks2[2][64 * HD_];   // [key][d]   2 x 16 KB
  __shared__ __align__(16) __bf16 Vs2[2][HD_ * 64];   // [d][key]   2 x 16 KB
  __shared__ __align__(16) __bf16 Ps[8 * 16 * 64];    // per-wave P 16 KB

  const int tid  = threadIdx.x;
  const int lane = tid & 31;
  const int wave = tid >> 5;
  const int l15  = lane & 15;
  const int roff = (lane < 16) ? 0 : 8;
  const int klo  = (lane < 16) ? 0 : 8;
  const int kb   = (lane < 16) ? 0 : 8;

  const int b  = blockIdx.z;
  const int h  = blockIdx.y;
  const int q0 = blockIdx.x * 128;

  const __bf16* Qb  = Q   + (((size_t)b * NQ_  + h)        * S_) * (size_t)HD_;
  const __bf16* Kb  = Kh  + (((size_t)b * NKV_ + (h >> 2)) * S_) * (size_t)HD_;
  const __bf16* VTb = VTh + (((size_t)b * NKV_ + (h >> 2)) * (size_t)HD_) * S_;

  // 8 async 16B copies per thread per tile (4 for K, 4 for V^T).
  auto issue = [&](int j0, int bi) {
#pragma unroll
    for (int it = 0; it < 4; ++it) {
      int idx = tid + it * 256;           // 0..1023
      int key = idx >> 4;
      int c8  = (idx & 15) << 3;
      cp16(&Ks2[bi][key * HD_ + c8], Kb + (size_t)(j0 + key) * HD_ + c8);
    }
#pragma unroll
    for (int it = 0; it < 4; ++it) {
      int idx = tid + it * 256;           // 0..1023
      int d   = idx >> 3;
      int c8  = (idx & 7) << 3;
      cp16(&Vs2[bi][d * 64 + c8], VTb + (size_t)d * S_ + j0 + c8);
    }
  };

  // Q fragments for this wave's 16 rows, resident for the whole loop.
  const int qrow = q0 + wave * 16 + l15;
  bf16x16 qf[4];
#pragma unroll
  for (int kc = 0; kc < 4; ++kc) {
    const __bf16* p = Qb + (size_t)qrow * HD_ + kc * 32;
    qf[kc] = cat8(*(const bf16x8*)(p + klo), *(const bf16x8*)(p + klo + 16));
  }

  float mrow[8], lrow[8];
  f32x8 oacc[8] = {};
#pragma unroll
  for (int r = 0; r < 8; ++r) { mrow[r] = -1e30f; lrow[r] = 0.f; }

  const int   ntiles = (q0 + 128) / 64;       // keys 0 .. q0+127
  const float scale  = 0.08838834764831845f;  // 1/sqrt(128)

  issue(0, 0);
  for (int jt = 0; jt < ntiles; ++jt) {
    const int j0  = jt * 64;
    const int cur = jt & 1;
    if (jt + 1 < ntiles) {
      issue((jt + 1) * 64, cur ^ 1);
      ASYNC_WAIT(8);
    } else {
      ASYNC_WAIT(0);
    }
    __syncthreads();

    const __bf16* Ksb = Ks2[cur];
    const __bf16* Vsb = Vs2[cur];

    // Scores: 16 x 64 = 4 fragments, K accumulated over HD=128 (4 x WMMA).
    f32x8 sf[4] = {};
#pragma unroll
    for (int ni = 0; ni < 4; ++ni) {
      const int j = ni * 16 + l15;  // key column (B-fragment lane = column)
#pragma unroll
      for (int kc = 0; kc < 4; ++kc) {
        const __bf16* p = Ksb + j * HD_ + kc * 32 + kb;
        bf16x16 kf = cat8(*(const bf16x8*)p, *(const bf16x8*)(p + 16));
        sf[ni] = wmma_bf16(qf[kc], kf, sf[ni]);
      }
    }

    // Online softmax: rows r live in half-wave groups; reduce over 16 lanes.
#pragma unroll
    for (int r = 0; r < 8; ++r) {
      const int qr = q0 + wave * 16 + roff + r;
      float mx = -1e30f;
#pragma unroll
      for (int ni = 0; ni < 4; ++ni) {
        float s = sf[ni][r] * scale;
        if (j0 + ni * 16 + l15 > qr) s = -1e30f;  // causal mask
        sf[ni][r] = s;
        mx = fmaxf(mx, s);
      }
#pragma unroll
      for (int mk = 1; mk < 16; mk <<= 1) mx = fmaxf(mx, __shfl_xor(mx, mk, 32));
      const float mnew  = fmaxf(mrow[r], mx);
      const float alpha = __expf(mrow[r] - mnew);
      float sum = 0.f;
#pragma unroll
      for (int ni = 0; ni < 4; ++ni) {
        float p = __expf(sf[ni][r] - mnew);
        sf[ni][r] = p;
        sum += p;
      }
#pragma unroll
      for (int mk = 1; mk < 16; mk <<= 1) sum += __shfl_xor(sum, mk, 32);
      lrow[r] = lrow[r] * alpha + sum;
      mrow[r] = mnew;
#pragma unroll
      for (int nb = 0; nb < 8; ++nb) oacc[nb][r] *= alpha;
    }

    // Stage P to per-wave LDS (D-fragment -> A-fragment layout bridge).
    __bf16* Pw = Ps + wave * 16 * 64;
#pragma unroll
    for (int ni = 0; ni < 4; ++ni)
#pragma unroll
      for (int r = 0; r < 8; ++r)
        Pw[(roff + r) * 64 + ni * 16 + l15] = (__bf16)sf[ni][r];

    // P(16x64) @ V(64x128): 2 K-chunks x 8 N-tiles.
    bf16x16 pf[2];
#pragma unroll
    for (int kc = 0; kc < 2; ++kc) {
      const __bf16* p = Pw + l15 * 64 + kc * 32;
      pf[kc] = cat8(*(const bf16x8*)(p + klo), *(const bf16x8*)(p + klo + 16));
    }
#pragma unroll
    for (int nb = 0; nb < 8; ++nb) {
      const int n = nb * 16 + l15;
#pragma unroll
      for (int kc = 0; kc < 2; ++kc) {
        const __bf16* p = Vsb + n * 64 + kc * 32 + kb;
        bf16x16 vf = cat8(*(const bf16x8*)p, *(const bf16x8*)(p + 16));
        oacc[nb] = wmma_bf16(pf[kc], vf, oacc[nb]);
      }
    }
    __syncthreads();
  }

  // Normalize and store O in x-layout [b*S, NQ*HD] (bf16) for the Wo GEMM.
#pragma unroll
  for (int r = 0; r < 8; ++r) {
    const float inv = (lrow[r] > 0.f) ? 1.f / lrow[r] : 0.f;
    const int q = q0 + wave * 16 + roff + r;
    __bf16* orow = O + ((size_t)b * S_ + q) * (size_t)D_ + h * HD_;
#pragma unroll
    for (int nb = 0; nb < 8; ++nb)
      orow[nb * 16 + l15] = (__bf16)(oacc[nb][r] * inv);
  }
}

// ---------------------------------------------------------------------------
// Launch
// ---------------------------------------------------------------------------
extern "C" void kernel_launch(void* const* d_in, const int* in_sizes, int n_in,
                              void* d_out, int out_size, void* d_ws, size_t ws_size,
                              hipStream_t stream) {
  const float* x  = (const float*)d_in[0];
  const float* fc = (const float*)d_in[1];  // (S, HD/2, 2) cos/sin
  const float* Wq = (const float*)d_in[2];
  const float* bq = (const float*)d_in[3];
  const float* Wk = (const float*)d_in[4];
  const float* bk = (const float*)d_in[5];
  const float* Wv = (const float*)d_in[6];
  const float* bv = (const float*)d_in[7];
  const float* Wo = (const float*)d_in[8];
  const int*   sp = (const int*)d_in[9];
  float* out = (float*)d_out;
  (void)in_sizes; (void)n_in; (void)out_size; (void)ws_size;

  const size_t M = (size_t)B_ * S_;  // 8192 tokens

  char* ws = (char*)d_ws;
  size_t off = 0;
  auto alloc = [&](size_t elems) {
    __bf16* p = (__bf16*)(ws + off);
    off += elems * 2;
    return p;
  };
  __bf16* xh  = alloc(M * D_);                    // x in bf16
  __bf16* wqh = alloc((size_t)D_ * D_);           // Wq^T [N][K]
  __bf16* wkh = alloc((size_t)D_ * NKV_ * HD_);   // Wk^T
  __bf16* wvh = alloc((size_t)D_ * NKV_ * HD_);   // Wv^T
  __bf16* woh = alloc((size_t)D_ * D_);           // Wo^T
  __bf16* qh  = alloc(M * D_);                    // Q [b,h,s,d]
  __bf16* kh  = alloc(M * NKV_ * HD_);            // K [b,kv,s,d]
  __bf16* vh  = alloc(M * NKV_ * HD_);            // V^T [b,kv,d,s]
  __bf16* oh  = alloc(M * D_);                    // attn out, x-layout

  f2bf_kernel<<<dim3((unsigned)((M * D_ + 255) / 256)), 256, 0, stream>>>(
      x, xh, (int)(M * D_));
  f2bfT_kernel<<<dim3((unsigned)(((size_t)D_ * D_ + 255) / 256)), 256, 0, stream>>>(
      Wq, wqh, D_, D_);
  f2bfT_kernel<<<dim3((unsigned)(((size_t)D_ * NKV_ * HD_ + 255) / 256)), 256, 0, stream>>>(
      Wk, wkh, D_, NKV_ * HD_);
  f2bfT_kernel<<<dim3((unsigned)(((size_t)D_ * NKV_ * HD_ + 255) / 256)), 256, 0, stream>>>(
      Wv, wvh, D_, NKV_ * HD_);
  f2bfT_kernel<<<dim3((unsigned)(((size_t)D_ * D_ + 255) / 256)), 256, 0, stream>>>(
      Wo, woh, D_, D_);

  // Q = rope(x@Wq + bq) -> [b, NQ, s, d] bf16
  gemm_bf16_kernel<<<dim3(D_ / GTN, (unsigned)(M / GTM)), 256, 0, stream>>>(
      xh, wqh, bq, fc, sp, nullptr, qh, (int)M, D_, D_, 1, NQ_);
  // K = rope(x@Wk + bk) -> [b, NKV, s, d] bf16
  gemm_bf16_kernel<<<dim3((NKV_ * HD_) / GTN, (unsigned)(M / GTM)), 256, 0, stream>>>(
      xh, wkh, bk, fc, sp, nullptr, kh, (int)M, NKV_ * HD_, D_, 1, NKV_);
  // V = x@Wv + bv -> [b, NKV, d, s] bf16 (transposed for flash async staging)
  gemm_bf16_kernel<<<dim3((NKV_ * HD_) / GTN, (unsigned)(M / GTM)), 256, 0, stream>>>(
      xh, wvh, bv, fc, sp, nullptr, vh, (int)M, NKV_ * HD_, D_, 3, NKV_);

  // Causal GQA flash attention -> oh (x-layout bf16)
  flash_kernel<<<dim3(S_ / 128, NQ_, B_), 256, 0, stream>>>(qh, kh, vh, oh);

  // out = oh @ Wo (fp32)
  gemm_bf16_kernel<<<dim3(D_ / GTN, (unsigned)(M / GTM)), 256, 0, stream>>>(
      oh, woh, nullptr, nullptr, nullptr, out, nullptr, (int)M, D_, D_, 0, 0);
}
